// Gpt2Attention_23433341567274
// MI455X (gfx1250) — matrix operand order repaired and checked by reading, not compiled, not run on hardware
//
#include <hip/hip_runtime.h>

// ---------------------------------------------------------------------------
// GPT-2 attention block for MI455X (gfx1250, wave32, WMMA).
// - All GEMMs: V_WMMA_F32_16X16X32_BF16 (bf16 in, f32 accumulate).
// - Attention is flash-style fused (scores written once, never re-read;
//   the 1.07 GB score tensor write is the roofline floor, ~50us @ 23.3 TB/s).
// - K/V tiles staged to LDS with GLOBAL_LOAD_ASYNC_TO_LDS_B128, double
//   buffered via ASYNCcnt, killing the 8x per-block redundant L2 reads.
// ---------------------------------------------------------------------------

#define H    1024
#define NH   16
#define HD   64
#define SEQ  2048
#define BATCH 4
#define MBIAS (-10000.0f)

#define KSTRIDE 72  // bf16 elems per K LDS row (64 data + 8 pad) = 144 B
#define VSTRIDE 40  // bf16 elems per V LDS row (32 data + 8 pad) = 80 B

typedef __bf16 bf16;
typedef bf16  v16bf __attribute__((ext_vector_type(16)));
typedef bf16  v8bf  __attribute__((ext_vector_type(8)));
typedef float v8f   __attribute__((ext_vector_type(8)));

static __device__ __forceinline__ v8f wmma_bf16(v16bf a, v16bf b, v8f c) {
  return __builtin_amdgcn_wmma_f32_16x16x32_bf16(
      /*neg_a=*/false, a, /*neg_b=*/false, b,
      /*c_mod=*/(short)0, c, /*reuse_a=*/false, /*reuse_b=*/false);
}

// A fragment (16x32, MxK): lanes 0-15 rows M=0..15, K in kh..kh+15,
// kh=(lane>>4)*16. p pre-offset by (row, k0+kh).
static __device__ __forceinline__ v16bf load_a16x32(const bf16* p) {
  v8bf lo = *(const v8bf*)p;
  v8bf hi = *(const v8bf*)(p + 8);
  v16bf r;
#pragma unroll
  for (int i = 0; i < 8; ++i) { r[i] = lo[i]; r[i + 8] = hi[i]; }
  return r;
}

// B fragment (32x16, KxN): lane = column, lane halves split K 0-15/16-31;
// 16 contiguous K values per lane. p = &Bcolmajor[col][k0 + (lane>>4)*16].
static __device__ __forceinline__ v16bf load_b32x16(const bf16* p) {
  v8bf lo = *(const v8bf*)p;
  v8bf hi = *(const v8bf*)(p + 8);
  v16bf r;
#pragma unroll
  for (int i = 0; i < 8; ++i) { r[i] = lo[i]; r[i + 8] = hi[i]; }
  return r;
}

// 16-byte async global->LDS copy (per-lane addresses, tracked by ASYNCcnt).
static __device__ __forceinline__ void async_copy16(void* lds_dst,
                                                    const void* gsrc) {
  unsigned ldsa = (unsigned)(uintptr_t)lds_dst;  // low 32 bits = LDS offset
  asm volatile("global_load_async_to_lds_b128 %0, %1, off"
               :: "v"(ldsa), "v"(gsrc)
               : "memory");
}
static __device__ __forceinline__ void wait_async2() {
  asm volatile("s_wait_asynccnt 0x2" ::: "memory");
}
static __device__ __forceinline__ void wait_async0() {
  asm volatile("s_wait_asynccnt 0x0" ::: "memory");
}

// ---------------------------------------------------------------------------
// LayerNorm -> bf16.  One block per row of 1024.
// ---------------------------------------------------------------------------
__global__ __launch_bounds__(256) void ln_kernel(const float* __restrict__ hs,
                                                 const float* __restrict__ g,
                                                 const float* __restrict__ be,
                                                 bf16* __restrict__ xb) {
  int row  = blockIdx.x;
  int tid  = threadIdx.x;
  int lane = tid & 31, wid = tid >> 5;
  const float* p = hs + (size_t)row * H;
  float v[4];
  float s = 0.f, q = 0.f;
#pragma unroll
  for (int i = 0; i < 4; ++i) {
    v[i] = p[tid + i * 256];
    s += v[i];
    q += v[i] * v[i];
  }
#pragma unroll
  for (int o = 16; o; o >>= 1) {
    s += __shfl_xor(s, o, 32);
    q += __shfl_xor(q, o, 32);
  }
  __shared__ float rs[8], rq[8];
  if (lane == 0) { rs[wid] = s; rq[wid] = q; }
  __syncthreads();
  float ts = 0.f, tq = 0.f;
#pragma unroll
  for (int w = 0; w < 8; ++w) { ts += rs[w]; tq += rq[w]; }
  float mean = ts * (1.0f / H);
  float var  = tq * (1.0f / H) - mean * mean;
  float rstd = rsqrtf(var + 1e-5f);
  bf16* o = xb + (size_t)row * H;
#pragma unroll
  for (int i = 0; i < 4; ++i) {
    int c = tid + i * 256;
    o[c] = (bf16)((v[i] - mean) * rstd * g[c] + be[c]);
  }
}

// ---------------------------------------------------------------------------
// fp32 [Kd,N] row-major -> bf16 [N,Kd] (column-major weights so WMMA B
// fragments are contiguous 32B-per-lane loads).
// ---------------------------------------------------------------------------
__global__ __launch_bounds__(256) void transpose_bf16_kernel(
    const float* __restrict__ in, bf16* __restrict__ out, int Kd, int N) {
  __shared__ float tile[32][33];
  int tx = threadIdx.x & 31, ty = threadIdx.x >> 5;  // 32 x 8
  int n0 = blockIdx.x * 32, k0 = blockIdx.y * 32;
#pragma unroll
  for (int i = 0; i < 32; i += 8)
    tile[ty + i][tx] = in[(size_t)(k0 + ty + i) * N + n0 + tx];
  __syncthreads();
#pragma unroll
  for (int i = 0; i < 32; i += 8)
    out[(size_t)(n0 + ty + i) * Kd + k0 + tx] = (bf16)tile[tx][ty + i];
}

// ---------------------------------------------------------------------------
// QKV GEMM: [8192,1024] x [1024,3072] + bias. Block 64x128 (8 waves 4x2),
// wave tile 16x64, K step 32. Epilogue: bf16 Q/K [B,NH,S,HD], V [B,NH,HD,S].
// ---------------------------------------------------------------------------
__global__ __launch_bounds__(256) void gemm_qkv_kernel(
    const bf16* __restrict__ A, const bf16* __restrict__ WT,
    const float* __restrict__ bias, bf16* __restrict__ Qb,
    bf16* __restrict__ Kb, bf16* __restrict__ Vb) {
  const int Kd = H;
  int lane = threadIdx.x & 31, wave = threadIdx.x >> 5;
  int rowBase = blockIdx.y * 64 + (wave >> 1) * 16;
  int colBase = blockIdx.x * 128 + (wave & 1) * 64;
  int m  = lane & 15;
  int kh = (lane >> 4) * 16;
  int kb = (lane >> 4) * 16;
  v8f acc[4];
#pragma unroll
  for (int t = 0; t < 4; ++t) acc[t] = (v8f)(0.0f);

  for (int k0 = 0; k0 < Kd; k0 += 32) {
    v16bf a = load_a16x32(A + (size_t)(rowBase + m) * Kd + k0 + kh);
#pragma unroll
    for (int t = 0; t < 4; ++t) {
      v16bf bfr = load_b32x16(WT + (size_t)(colBase + t * 16 + m) * Kd + k0 + kb);
      acc[t] = wmma_bf16(a, bfr, acc[t]);
    }
  }

  int rhalf = (lane >> 4) * 8;
#pragma unroll
  for (int t = 0; t < 4; ++t) {
    int col   = colBase + t * 16 + m;
    float bv  = bias[col];
    int which = col >> 10;  // 0=q 1=k 2=v
    int hh    = col & (H - 1);
    int head  = hh >> 6;
    int d     = hh & (HD - 1);
#pragma unroll
    for (int e = 0; e < 8; ++e) {
      int row = rowBase + rhalf + e;
      int bb  = row >> 11;
      int s   = row & (SEQ - 1);
      bf16 o  = (bf16)(acc[t][e] + bv);
      size_t sd  = (((size_t)bb * NH + head) * SEQ + s) * HD + d;
      size_t dsi = (((size_t)bb * NH + head) * HD + d) * SEQ + s;
      if (which == 0)      Qb[sd]  = o;
      else if (which == 1) Kb[sd]  = o;
      else                 Vb[dsi] = o;
    }
  }
}

// ---------------------------------------------------------------------------
// Fused attention. Per wave: 16 q-rows; k-loop in 32-col chunks.
// K/V chunk tiles staged to LDS via async copies (double buffered): one
// staging per block instead of per wave (8x less L2 read traffic).
// ---------------------------------------------------------------------------
__global__ __launch_bounds__(256) void attn_kernel(
    const bf16* __restrict__ Q, const bf16* __restrict__ K,
    const bf16* __restrict__ V, const float* __restrict__ amask,
    float* __restrict__ sout_base, bf16* __restrict__ ctx) {
  __shared__ __align__(16) bf16 Ktile[2][32 * KSTRIDE];  // 2 x 4608 B
  __shared__ __align__(16) bf16 Vtile[2][64 * VSTRIDE];  // 2 x 5120 B
  __shared__ __align__(16) bf16 P[8][16][40];            // probs, 10240 B

  int tid  = threadIdx.x;
  int lane = tid & 31, wave = tid >> 5;
  int qblk = blockIdx.x, h = blockIdx.y, b = blockIdx.z;
  int qbase = qblk * 128 + wave * 16;

  const bf16* Qh = Q + ((size_t)b * NH + h) * SEQ * HD;
  const bf16* Kh = K + ((size_t)b * NH + h) * SEQ * HD;
  const bf16* Vh = V + ((size_t)b * NH + h) * HD * SEQ;   // [HD][SEQ]
  float* sout    = sout_base + ((size_t)b * NH + h) * SEQ * SEQ;
  const float* am = amask + (size_t)b * SEQ;

  int m     = lane & 15;
  int half  = lane >> 4;
  int kh    = half * 16;
  int kb    = half * 16;
  int rhalf = half * 8;

  // staging assignments (all 256 threads): K = 32 rows x 8 segs of 16B,
  // V = 64 rows x 4 segs of 16B.
  int krow = tid >> 3, kseg = tid & 7;
  int vrow = tid >> 2, vseg = tid & 3;

  const bf16* qp = Qh + (size_t)(qbase + m) * HD + kh;
  v16bf aq0 = load_a16x32(qp);        // d = 0..31
  v16bf aq1 = load_a16x32(qp + 32);   // d = 32..63

  v8f acc[4];
#pragma unroll
  for (int t = 0; t < 4; ++t) acc[t] = (v8f)(0.0f);
  float rmax[8], rsum[8];
#pragma unroll
  for (int e = 0; e < 8; ++e) { rmax[e] = -1e30f; rsum[e] = 0.f; }

  int kend = qblk * 128 + 128;  // causal bound, uniform across block

  // prologue: stage chunk 0 into buffer 0
  async_copy16(&Ktile[0][krow * KSTRIDE + kseg * 8],
               Kh + (size_t)krow * HD + kseg * 8);
  async_copy16(&Vtile[0][vrow * VSTRIDE + vseg * 8],
               Vh + (size_t)vrow * SEQ + vseg * 8);

  int cur = 0;
  for (int k0 = 0; k0 < kend; k0 += 32) {
    int nxt = cur ^ 1;
    bool hasNext = (k0 + 32) < kend;
    if (hasNext) {
      int kn = k0 + 32;
      async_copy16(&Ktile[nxt][krow * KSTRIDE + kseg * 8],
                   Kh + (size_t)(kn + krow) * HD + kseg * 8);
      async_copy16(&Vtile[nxt][vrow * VSTRIDE + vseg * 8],
                   Vh + (size_t)vrow * SEQ + kn + vseg * 8);
      wait_async2();   // current chunk's 2 copies done; next stays in flight
    } else {
      wait_async0();
    }
    __syncthreads();

    const bf16* kt = Ktile[cur];
    const bf16* vt = Vtile[cur];

    // ---- scores: S = Q(16x64) @ K^T(64x32), two 16-col tiles ----
    const bf16* kp0 = kt + (size_t)m * KSTRIDE + kb;           // rows 0..15
    const bf16* kp1 = kp0 + 16 * KSTRIDE;                      // rows 16..31
    v16bf b00 = load_b32x16(kp0);        // d = kb..kb+15
    v16bf b01 = load_b32x16(kp0 + 32);   // d = 32+kb..
    v16bf b10 = load_b32x16(kp1);
    v16bf b11 = load_b32x16(kp1 + 32);
    v8f s0 = (v8f)(0.0f), s1 = (v8f)(0.0f);
    s0 = wmma_bf16(aq0, b00, s0);
    s0 = wmma_bf16(aq1, b01, s0);
    s1 = wmma_bf16(aq0, b10, s1);
    s1 = wmma_bf16(aq1, b11, s1);

    int col0 = k0 + m, col1 = col0 + 16;
    float am0 = am[col0] * MBIAS;
    float am1 = am[col1] * MBIAS;

#pragma unroll
    for (int e = 0; e < 8; ++e) {
      int qr = qbase + rhalf + e;
      float v0 = (col0 <= qr ? s0[e] * 0.125f : MBIAS) + am0;
      float v1 = (col1 <= qr ? s1[e] * 0.125f : MBIAS) + am1;
      sout[(size_t)qr * SEQ + col0] = v0;   // masked pre-softmax scores out
      sout[(size_t)qr * SEQ + col1] = v1;

      float mx = fmaxf(v0, v1);
#pragma unroll
      for (int o = 1; o < 16; o <<= 1) mx = fmaxf(mx, __shfl_xor(mx, o, 32));
      float nm   = fmaxf(rmax[e], mx);
      float corr = __expf(rmax[e] - nm);
      rmax[e] = nm;
      float p0 = __expf(v0 - nm), p1 = __expf(v1 - nm);
      float ps = p0 + p1;
#pragma unroll
      for (int o = 1; o < 16; o <<= 1) ps += __shfl_xor(ps, o, 32);
      rsum[e] = rsum[e] * corr + ps;
      acc[0][e] *= corr; acc[1][e] *= corr;
      acc[2][e] *= corr; acc[3][e] *= corr;
      P[wave][rhalf + e][m]      = (bf16)p0;
      P[wave][rhalf + e][16 + m] = (bf16)p1;
    }
    __syncthreads();

    // ---- ctx += P(16x32) @ V(32x64) ----
    v16bf ap = load_a16x32(&P[wave][m][kh]);
#pragma unroll
    for (int t = 0; t < 4; ++t) {
      const bf16* vp = vt + (size_t)(t * 16 + m) * VSTRIDE + kb;
      v16bf bv = load_b32x16(vp);
      acc[t] = wmma_bf16(ap, bv, acc[t]);
    }
    __syncthreads();  // protects P and the buffer being re-staged next iter
    cur = nxt;
  }

  // ---- fill fully-masked upper-triangle remainder of scores (b128) ----
  for (int rr = 0; rr < 16; ++rr) {
    int qr = qbase + rr;
    float* rowp = sout + (size_t)qr * SEQ;
    for (int c = kend + lane * 4; c < SEQ; c += 128) {
      float4 a4 = *(const float4*)(am + c);
      float4 w;
      w.x = MBIAS + a4.x * MBIAS;
      w.y = MBIAS + a4.y * MBIAS;
      w.z = MBIAS + a4.z * MBIAS;
      w.w = MBIAS + a4.w * MBIAS;
      *(float4*)(rowp + c) = w;
    }
  }

  // ---- normalize and write ctx (bf16, [B,S,H]) ----
#pragma unroll
  for (int t = 0; t < 4; ++t) {
#pragma unroll
    for (int e = 0; e < 8; ++e) {
      int row = qbase + rhalf + e;
      int d   = t * 16 + m;
      ctx[((size_t)b * SEQ + row) * H + h * HD + d] =
          (bf16)(acc[t][e] / rsum[e]);
    }
  }
}

// ---------------------------------------------------------------------------
// Projection GEMM: out = hidden + ctx @ Wp + bp   (fp32 out)
// ---------------------------------------------------------------------------
__global__ __launch_bounds__(256) void gemm_proj_kernel(
    const bf16* __restrict__ A, const bf16* __restrict__ WT,
    const float* __restrict__ bias, const float* __restrict__ resid,
    float* __restrict__ out) {
  const int Kd = H;
  int lane = threadIdx.x & 31, wave = threadIdx.x >> 5;
  int rowBase = blockIdx.y * 64 + (wave >> 1) * 16;
  int colBase = blockIdx.x * 128 + (wave & 1) * 64;
  int m  = lane & 15;
  int kh = (lane >> 4) * 16;
  int kb = (lane >> 4) * 16;
  v8f acc[4];
#pragma unroll
  for (int t = 0; t < 4; ++t) acc[t] = (v8f)(0.0f);

  for (int k0 = 0; k0 < Kd; k0 += 32) {
    v16bf a = load_a16x32(A + (size_t)(rowBase + m) * Kd + k0 + kh);
#pragma unroll
    for (int t = 0; t < 4; ++t) {
      v16bf bfr = load_b32x16(WT + (size_t)(colBase + t * 16 + m) * Kd + k0 + kb);
      acc[t] = wmma_bf16(a, bfr, acc[t]);
    }
  }

  int rhalf = (lane >> 4) * 8;
#pragma unroll
  for (int t = 0; t < 4; ++t) {
    int col  = colBase + t * 16 + m;
    float bv = bias[col];
#pragma unroll
    for (int e = 0; e < 8; ++e) {
      int row = rowBase + rhalf + e;
      size_t idx = (size_t)row * H + col;
      out[idx] = acc[t][e] + bv + resid[idx];
    }
  }
}

// ---------------------------------------------------------------------------
extern "C" void kernel_launch(void* const* d_in, const int* in_sizes, int n_in,
                              void* d_out, int out_size, void* d_ws,
                              size_t ws_size, hipStream_t stream) {
  (void)in_sizes; (void)n_in; (void)out_size; (void)ws_size;
  const float* hs       = (const float*)d_in[0];
  const float* amask    = (const float*)d_in[1];
  const float* c_attn_w = (const float*)d_in[2];
  const float* c_attn_b = (const float*)d_in[3];
  const float* c_proj_w = (const float*)d_in[4];
  const float* c_proj_b = (const float*)d_in[5];
  const float* ln_g     = (const float*)d_in[6];
  const float* ln_b     = (const float*)d_in[7];

  float* out    = (float*)d_out;
  float* scores = out + (size_t)BATCH * SEQ * H;  // second tuple element

  bf16* ws = (bf16*)d_ws;
  size_t o = 0;
  bf16* xb   = ws + o; o += (size_t)BATCH * SEQ * H;  // LN output
  bf16* wTA  = ws + o; o += (size_t)3 * H * H;        // c_attn_w^T
  bf16* wTP  = ws + o; o += (size_t)H * H;            // c_proj_w^T
  bf16* Qb   = ws + o; o += (size_t)BATCH * SEQ * H;  // [B,NH,S,HD]
  bf16* Kb   = ws + o; o += (size_t)BATCH * SEQ * H;  // [B,NH,S,HD]
  bf16* Vb   = ws + o; o += (size_t)BATCH * SEQ * H;  // [B,NH,HD,S]
  bf16* ctxb = ws + o; o += (size_t)BATCH * SEQ * H;  // [B,S,H]

  ln_kernel<<<BATCH * SEQ, 256, 0, stream>>>(hs, ln_g, ln_b, xb);
  transpose_bf16_kernel<<<dim3(3 * H / 32, H / 32), 256, 0, stream>>>(
      c_attn_w, wTA, H, 3 * H);
  transpose_bf16_kernel<<<dim3(H / 32, H / 32), 256, 0, stream>>>(
      c_proj_w, wTP, H, H);
  gemm_qkv_kernel<<<dim3(3 * H / 128, BATCH * SEQ / 64), 256, 0, stream>>>(
      xb, wTA, c_attn_b, Qb, Kb, Vb);
  attn_kernel<<<dim3(SEQ / 128, NH, BATCH), 256, 0, stream>>>(
      Qb, Kb, Vb, amask, scores, ctxb);
  gemm_proj_kernel<<<dim3(H / 128, BATCH * SEQ / 64), 256, 0, stream>>>(
      ctxb, wTP, c_proj_b, hs, out);
}
